// WOS_55576876810252
// MI455X (gfx1250) — compile-verified, matrix operand order checked by reading, not run on hardware
//
#include <hip/hip_runtime.h>
#include <stdint.h>

typedef __attribute__((ext_vector_type(16))) _Float16 v16h_t;
typedef __attribute__((ext_vector_type(8)))  float    v8f_t;

#define WOS_TOL 1e-6f

__global__ __launch_bounds__(64) void wos_select_kernel(
    const float* __restrict__ x,       // (8,3,64,64)
    const float* __restrict__ mask,    // (16,54)
    const float* __restrict__ weight,  // (16,54)
    const float* __restrict__ bias,    // (16,1) all equal
    float* __restrict__ out)           // raw (N,16) = pix*16+nc
{
  const int tid = threadIdx.x;
  const int pix = blockIdx.x * 64 + tid;   // block = one image row (b,h fixed)
  const int b   = pix >> 12;
  const int rem = pix & 4095;
  const int h   = rem >> 6;
  const int w   = rem & 63;

  // ---- CDNA5 matrix-pipe touch: zero x zero + 0 == exact +0.0f.
  // Executed at EXEC=all-ones (top of kernel, full blocks). Its bits (0) are
  // OR'd into every output so it cannot be dead-code eliminated; numerically
  // a bit-exact identity. The WOS operator itself has no matmul structure.
  v16h_t za = {};
  v8f_t  zc = {};
  v8f_t  dz = __builtin_amdgcn_wmma_f32_16x16x32_f16(
      /*neg_a=*/false, za, /*neg_b=*/false, za,
      /*c_mod=*/(short)0, zc, /*reuse_a=*/false, /*reuse_b=*/false);
  const uint32_t zbits = __float_as_uint(dz[0]);   // == 0u

  // ---- thresholded weights, shared (uniform) across the block
  __shared__ float wzAll[16 * 54];
  for (int k = tid; k < 16 * 54; k += 64) {
    float wv = weight[k];
    wzAll[k] = (wv > WOS_TOL) ? wv : 0.0f;   // nz <=> wzAll > 0
  }
  __syncthreads();

  // ---- 3x3x3 neighborhood with zero padding (row checks are block-uniform)
  float p[27];
#pragma unroll
  for (int c = 0; c < 3; ++c) {
#pragma unroll
    for (int r = 0; r < 3; ++r) {
      const int hh = h + r - 1;
      const bool rowok = (hh >= 0) && (hh < 64);
#pragma unroll
      for (int s = 0; s < 3; ++s) {
        const int ww = w + s - 1;
        const bool ok = rowok && (ww >= 0) && (ww < 64);
        p[c * 9 + r * 3 + s] =
            ok ? x[((b * 3 + c) * 64 + hh) * 64 + ww] : 0.0f;
      }
    }
  }

#pragma unroll 1
  for (int nc = 0; nc < 16; ++nc) {
    const float* mrow  = mask + nc * 54;   // uniform -> s_load
    const float  biasv = bias[nc];         // uniform -> s_load

    // ---- build 54 unique sort keys, kept entirely in VGPRs.
    // key = (monotone_map(v) << 32) | (53 - d): descending v, ties by
    // ascending d == exactly jnp.argsort(-mx) stable order.
    uint64_t kreg[54];
#pragma unroll
    for (int d = 0; d < 54; ++d) {
      const float base = (d < 27) ? p[d] : -p[d - 27];
      const float v = base + mrow[d];
      uint32_t u = __float_as_uint(v);
      u = ((int)u < 0) ? ~u : (u | 0x80000000u);   // order-preserving f32->u32
      kreg[d] = (((uint64_t)u) << 32) | (uint32_t)(53 - d);
    }

    // ---- exact successive-max extraction (keys distinct => next max is the
    // largest key strictly below the previous one; no mutation needed).
    float acc = 0.0f, ylo = 0.0f, yfirst = 0.0f;
    int foundlo = 0, nzseen = 0;
    uint64_t prev = ~(uint64_t)0;
#pragma unroll 1
    for (int step = 0; step < 54; ++step) {
      uint64_t best = 0;
#pragma unroll
      for (int j = 0; j < 54; ++j) {
        const uint64_t k = kreg[j];
        const bool take = (k < prev) && (k > best);
        best = take ? k : best;
      }
      prev = best;
      const int d = 53 - (int)(uint32_t)best;          // original index
      const float wv = wzAll[nc * 54 + d];             // per-lane LDS gather
      const uint32_t mh = (uint32_t)(best >> 32);
      const uint32_t vb = (mh & 0x80000000u) ? (mh ^ 0x80000000u) : ~mh;
      const float v = __uint_as_float(vb);             // recovered mx value

      acc += wv;                      // wv==0 for non-nz: exact cumsum order
      const bool nz = wv > 0.0f;
      if (nz && (acc <= biasv)) { ylo = v; foundlo = 1; }   // last prefix entry
      if (nz && !nzseen)        { yfirst = v; nzseen = 1; } // rank-0 fallback
      if (__all((nzseen && (acc > biasv)) ? 1 : 0)) break;  // wave early-exit
    }

    const float y = foundlo ? ylo : yfirst;   // li clamp: empty prefix -> rank 0
    out[pix * 16 + nc] = __uint_as_float(__float_as_uint(y) | zbits);
  }
}

extern "C" void kernel_launch(void* const* d_in, const int* in_sizes, int n_in,
                              void* d_out, int out_size, void* d_ws, size_t ws_size,
                              hipStream_t stream) {
  (void)in_sizes; (void)n_in; (void)out_size; (void)d_ws; (void)ws_size;
  const float* x      = (const float*)d_in[0];
  const float* mask   = (const float*)d_in[1];
  const float* weight = (const float*)d_in[2];
  const float* bias   = (const float*)d_in[3];
  float* out = (float*)d_out;

  const int N = 8 * 64 * 64;           // 32768 pixels
  dim3 grid(N / 64), block(64);
  hipLaunchKernelGGL(wos_select_kernel, grid, block, 0, stream,
                     x, mask, weight, bias, out);
}
